// GNN_node_83468394430632
// MI455X (gfx1250) — matrix operand order repaired and checked by reading, not compiled, not run on hardware
//
#include <hip/hip_runtime.h>
#include <hip/hip_bf16.h>
#include <cstdint>

// ---------------------------------------------------------------------------
// GIN-style GNN forward for MI455X (gfx1250, wave32, WMMA).
//
// Roofline: ~27 GFLOP total; traffic dominated by L2-resident edge gathers
// (h = 20.5 MB << 192 MB L2) and 82M f32 scatter-atomics per layer. Dense
// GEMMs use v_wmma_f32_16x16x32_f16, fully unrolled K, b128 A loads,
// weights pre-packed into WMMA B-fragment order, and NTPW n-tiles per wave
// so each A fragment is loaded once and reused across NTPW WMMAs
// (cuts A-operand L2 traffic 2-4x vs one-tile-per-wave).
// ---------------------------------------------------------------------------

typedef __attribute__((ext_vector_type(16))) _Float16 v16h;
typedef __attribute__((ext_vector_type(8)))  float    v8f;

#define DIMD     128
#define EDIMD    16
#define NNODES   40000
#define NEDGES   640000
#define NLAYERS  3

// ---------------------------------------------------------------------------
// Pack W[K,Nc] (f32, row-major) into f16 WMMA B-fragment order:
//   frag[(((nt*(K/32))+kt)*32 + lane)*16 + i] = W[kt*32 + 8*(lane>>4) + (i&7) + 16*(i>>3)][nt*16 + (lane&15)]
// so each lane's 16-half fragment is one contiguous, 32B-aligned run.
// ---------------------------------------------------------------------------
__global__ __launch_bounds__(256)
void pack_w_f16(const float* __restrict__ W, _Float16* __restrict__ dst,
                int K, int Nc)
{
    const int idx = blockIdx.x * blockDim.x + threadIdx.x;
    if (idx >= K * Nc) return;
    const int i    = idx & 15;
    const int lane = (idx >> 4) & 31;
    const int kt   = (idx >> 9) % (K / 32);
    const int nt   = (idx >> 9) / (K / 32);
    const int k    = kt * 32 + (lane >> 4) * 8 + (i & 7) + (i >> 3) * 16;
    const int n    = nt * 16 + (lane & 15);
    dst[idx] = (_Float16)W[(size_t)k * Nc + n];
}

// ---------------------------------------------------------------------------
// C[M,Nc] = act( bn_scale * (A@W + lin_b) + bn_shift )
//   FUSE: A = (1+eps[layer])*A0 + A1   (GIN combine), else A = A0.
// One wave computes a 16 x (16*NTPW) strip: A fragment loaded once per
// k-tile, reused across NTPW WMMAs. 4 waves/block; grid covers M/16 x 1.
// A 16x32 f16 layout: lane(l&15)=row; elems {0..7}=kb+0..7, {8..15}=kb+16..23,
//                     kb = 8*(l>>4)  -> two aligned float4 pairs per ktile.
// B read from pre-packed fragments (contiguous v16h per lane, 2x b128).
// C/D: vgpr v, lane l -> M = v + 8*(l>>4), N = l&15.
// ---------------------------------------------------------------------------
template<int K, int NTPW, bool FUSE>
__global__ __launch_bounds__(128)
void gemm_bias_bn_wmma(const float* __restrict__ A0,
                       const float* __restrict__ A1,
                       const float* __restrict__ epsv, int layer,
                       const _Float16* __restrict__ Wp,  // packed fragments
                       const float* __restrict__ lin_b,  // [Nc]
                       const float* __restrict__ bn_g,   // [Nc] or null
                       const float* __restrict__ bn_b,   // [Nc] or null
                       float bn_inv,
                       float* __restrict__ C,
                       int Nc, int do_relu)
{
    constexpr int KT = K / 32;
    const int lane    = threadIdx.x & 31;
    const int wave    = threadIdx.x >> 5;
    const int mBase   = blockIdx.x * 16;
    const int nTile0  = (blockIdx.y * 4 + wave) * NTPW;
    const int halfSel = lane >> 4;
    const int l15     = lane & 15;

    const int mA = mBase + l15;

    const float* arow0 = A0 + (size_t)mA * K;
    const float* arow1 = FUSE ? (A1 + (size_t)mA * K) : nullptr;
    const float  alpha = FUSE ? (1.0f + epsv[layer]) : 1.0f;

    v8f acc[NTPW] = {};

#pragma unroll
    for (int kt = 0; kt < KT; ++kt) {
        const int kbA = kt * 32 + halfSel * 8;
        // ---- A fragment: 4x b128 loads (+4 more when fusing agg) ----
        const float4 a0 = *(const float4*)(arow0 + kbA);
        const float4 a1 = *(const float4*)(arow0 + kbA + 4);
        const float4 a2 = *(const float4*)(arow0 + kbA + 16);
        const float4 a3 = *(const float4*)(arow0 + kbA + 20);
        float va[16] = { a0.x, a0.y, a0.z, a0.w,  a1.x, a1.y, a1.z, a1.w,
                         a2.x, a2.y, a2.z, a2.w,  a3.x, a3.y, a3.z, a3.w };
        if (FUSE) {
            const float4 g0 = *(const float4*)(arow1 + kbA);
            const float4 g1 = *(const float4*)(arow1 + kbA + 4);
            const float4 g2 = *(const float4*)(arow1 + kbA + 16);
            const float4 g3 = *(const float4*)(arow1 + kbA + 20);
            const float gg[16] = { g0.x, g0.y, g0.z, g0.w,  g1.x, g1.y, g1.z, g1.w,
                                   g2.x, g2.y, g2.z, g2.w,  g3.x, g3.y, g3.z, g3.w };
#pragma unroll
            for (int i = 0; i < 16; ++i) va[i] = alpha * va[i] + gg[i];
        }
        v16h af;
#pragma unroll
        for (int i = 0; i < 16; ++i) af[i] = (_Float16)va[i];

        // ---- NTPW pre-packed B fragments, one WMMA each (A reused) ----
#pragma unroll
        for (int t = 0; t < NTPW; ++t) {
            const v16h bf = *(const v16h*)(
                Wp + (((size_t)(nTile0 + t) * KT + kt) * 32 + lane) * 16);
            acc[t] = __builtin_amdgcn_wmma_f32_16x16x32_f16(
                         false, af, false, bf, (short)0, acc[t], false, false);
        }
    }

    // ---- fused epilogue: +bias, eval-BN, ReLU ----
#pragma unroll
    for (int t = 0; t < NTPW; ++t) {
        const int nB = (nTile0 + t) * 16 + l15;
        const float lb = lin_b[nB];
        float scl = 1.0f, sft = 0.0f;
        if (bn_g) { scl = bn_g[nB] * bn_inv; sft = bn_b[nB]; }
#pragma unroll
        for (int v = 0; v < 8; ++v) {
            const int m = mBase + v + halfSel * 8;
            float val = acc[t][v] + lb;
            val = scl * val + sft;
            if (do_relu) val = fmaxf(val, 0.0f);
            C[(size_t)m * Nc + nB] = val;  // 64B-contiguous per half-wave
        }
    }
}

// ---------------------------------------------------------------------------
// One wave per edge; lane handles 4 dims (coalesced 512B float4 gathers).
// Fuses edge embed (K=16 recompute), attention (xor-shuffle + v_exp),
// ReLU message, global_atomic_add_f32 scatter.
// ---------------------------------------------------------------------------
__global__ __launch_bounds__(256)
void edge_message(const float* __restrict__ h,
                  const float* __restrict__ edge_attr, // [E,16]
                  const float* __restrict__ eW,        // [16,128]
                  const float* __restrict__ eb,        // [128]
                  const float* __restrict__ aW,        // [256]
                  const float* __restrict__ ab,        // [1]
                  const int*   __restrict__ srcIdx,
                  const int*   __restrict__ dstIdx,
                  float* __restrict__ agg,
                  int E)
{
    const int gwave = (int)((blockIdx.x * (unsigned)blockDim.x + threadIdx.x) >> 5);
    const int lane  = threadIdx.x & 31;
    if (gwave >= E) return;  // uniform per wave

    const int s  = srcIdx[gwave];
    const int d  = dstIdx[gwave];
    const int n0 = lane * 4;

    const float4 vj = *(const float4*)(h + (size_t)s * DIMD + n0);
    const float4 vi = *(const float4*)(h + (size_t)d * DIMD + n0);
    const float4 wi = *(const float4*)(aW + n0);
    const float4 wj = *(const float4*)(aW + DIMD + n0);

    float part = vi.x * wi.x + vi.y * wi.y + vi.z * wi.z + vi.w * wi.w
               + vj.x * wj.x + vj.y * wj.y + vj.z * wj.z + vj.w * wj.w;
#pragma unroll
    for (int off = 16; off >= 1; off >>= 1)
        part += __shfl_xor(part, off);
    const float att = 1.0f / (1.0f + __expf(-(part + ab[0])));

    // edge embedding recomputed on the fly (replicated load, L1-served)
    const float eav = edge_attr[(size_t)gwave * EDIMD + (lane & 15)];
    float4 e = *(const float4*)(eb + n0);
#pragma unroll
    for (int k = 0; k < EDIMD; ++k) {
        const float  ek   = __shfl(eav, k);
        const float4 wrow = *(const float4*)(eW + (size_t)k * DIMD + n0);
        e.x += ek * wrow.x; e.y += ek * wrow.y;
        e.z += ek * wrow.z; e.w += ek * wrow.w;
    }

    float* out = agg + (size_t)d * DIMD + n0;
    unsafeAtomicAdd(out + 0, fmaxf(vj.x * att + e.x, 0.0f));
    unsafeAtomicAdd(out + 1, fmaxf(vj.y * att + e.y, 0.0f));
    unsafeAtomicAdd(out + 2, fmaxf(vj.z * att + e.z, 0.0f));
    unsafeAtomicAdd(out + 3, fmaxf(vj.w * att + e.w, 0.0f));
}

__global__ __launch_bounds__(256)
void zero_f4(float4* __restrict__ p, int n4)
{
    const int i = blockIdx.x * blockDim.x + threadIdx.x;
    if (i < n4) p[i] = make_float4(0.0f, 0.0f, 0.0f, 0.0f);
}

// ---------------------------------------------------------------------------
extern "C" void kernel_launch(void* const* d_in, const int* in_sizes, int n_in,
                              void* d_out, int out_size, void* d_ws, size_t ws_size,
                              hipStream_t stream)
{
    (void)in_sizes; (void)n_in; (void)out_size; (void)ws_size;

    const float* x         = (const float*)d_in[0];
    const float* edge_attr = (const float*)d_in[1];
    const float* node_W    = (const float*)d_in[2];
    const float* node_b    = (const float*)d_in[3];
    const float* edge_W    = (const float*)d_in[4];   // [L,16,128]
    const float* edge_b    = (const float*)d_in[5];   // [L,128]
    const float* attn_W    = (const float*)d_in[6];   // [L,256,1]
    const float* attn_b    = (const float*)d_in[7];   // [L,1]
    const float* eps       = (const float*)d_in[8];   // [L]
    const float* W1        = (const float*)d_in[9];   // [L,128,256]
    const float* b1        = (const float*)d_in[10];  // [L,256]
    const float* bn1_g     = (const float*)d_in[11];  // [L,256]
    const float* bn1_b     = (const float*)d_in[12];  // [L,256]
    const float* W2        = (const float*)d_in[13];  // [L,256,128]
    const float* b2        = (const float*)d_in[14];  // [L,128]
    const float* bn_g      = (const float*)d_in[15];  // [L,128]
    const float* bn_b      = (const float*)d_in[16];  // [L,128]
    const int*   edge_idx  = (const int*)d_in[17];    // [2,E]
    const int*   srcIdx    = edge_idx;
    const int*   dstIdx    = edge_idx + NEDGES;

    float* out = (float*)d_out;

    // workspace: h | h_next | agg | z1(256-wide) | packed f16 weights
    const size_t nodeBytes = (size_t)NNODES * DIMD * sizeof(float);
    char*      ws    = (char*)d_ws;
    float*     h     = (float*)(ws);
    float*     hn    = (float*)(ws + nodeBytes);
    float*     agg   = (float*)(ws + 2 * nodeBytes);
    float*     z1    = (float*)(ws + 3 * nodeBytes);          // 2*nodeBytes wide
    _Float16*  pNode = (_Float16*)(ws + 5 * nodeBytes);       // 128*128
    _Float16*  pW1   = pNode + DIMD * DIMD;                   // 3 x 128*256
    _Float16*  pW2   = pW1 + (size_t)NLAYERS * DIMD * 2 * DIMD; // 3 x 256*128

    const float bn_inv = 0.99999500003749968f;  // 1/sqrt(1 + 1e-5)
    const dim3 blkG(128);

    // ---- pre-pack weights into WMMA B-fragment order (one-shot, tiny) ----
    pack_w_f16<<<(DIMD * DIMD + 255) / 256, 256, 0, stream>>>(
        node_W, pNode, DIMD, DIMD);
    for (int l = 0; l < NLAYERS; ++l) {
        pack_w_f16<<<(DIMD * 2 * DIMD + 255) / 256, 256, 0, stream>>>(
            W1 + (size_t)l * DIMD * 2 * DIMD,
            pW1 + (size_t)l * DIMD * 2 * DIMD, DIMD, 2 * DIMD);
        pack_w_f16<<<(2 * DIMD * DIMD + 255) / 256, 256, 0, stream>>>(
            W2 + (size_t)l * 2 * DIMD * DIMD,
            pW2 + (size_t)l * 2 * DIMD * DIMD, 2 * DIMD, DIMD);
    }

    // ---- node encoder: h = x @ node_W + node_b  (K=128, Nc=128, NTPW=2) ----
    gemm_bias_bn_wmma<DIMD, 2, false>
        <<<dim3(NNODES / 16, 1), blkG, 0, stream>>>(
        x, nullptr, eps, 0, pNode, node_b, nullptr, nullptr, 1.0f,
        h, DIMD, 0);

    for (int l = 0; l < NLAYERS; ++l) {
        const int n4 = NNODES * DIMD / 4;
        zero_f4<<<(n4 + 255) / 256, 256, 0, stream>>>((float4*)agg, n4);

        const int edgeThreads = NEDGES * 32;
        edge_message<<<(edgeThreads + 255) / 256, 256, 0, stream>>>(
            h, edge_attr,
            edge_W + (size_t)l * EDIMD * DIMD, edge_b + (size_t)l * DIMD,
            attn_W + (size_t)l * 2 * DIMD,     attn_b + l,
            srcIdx, dstIdx, agg, NEDGES);

        // MLP1: z1 = relu(BN1(((1+eps)h + agg) @ W1 + b1))  K=128, Nc=256, NTPW=4
        gemm_bias_bn_wmma<DIMD, 4, true>
            <<<dim3(NNODES / 16, 1), blkG, 0, stream>>>(
            h, agg, eps, l,
            pW1 + (size_t)l * DIMD * 2 * DIMD, b1 + (size_t)l * 2 * DIMD,
            bn1_g + (size_t)l * 2 * DIMD, bn1_b + (size_t)l * 2 * DIMD, bn_inv,
            z1, 2 * DIMD, 1);

        // MLP2: h' = [relu(] BN(z1 @ W2 + b2) [)]           K=256, Nc=128, NTPW=2
        float* dstbuf = (l == NLAYERS - 1) ? out : hn;
        gemm_bias_bn_wmma<2 * DIMD, 2, false>
            <<<dim3(NNODES / 16, 1), blkG, 0, stream>>>(
            z1, nullptr, eps, l,
            pW2 + (size_t)l * 2 * DIMD * DIMD, b2 + (size_t)l * DIMD,
            bn_g + (size_t)l * DIMD, bn_b + (size_t)l * DIMD, bn_inv,
            dstbuf, DIMD, (l < NLAYERS - 1) ? 1 : 0);

        float* t = h; h = hn; hn = t;
    }
}
